// TransformerLayerMoE_30459908063736
// MI455X (gfx1250) — matrix operand order repaired and checked by reading, not compiled, not run on hardware
//
#include <hip/hip_runtime.h>

// ---------------------------------------------------------------------------
// CDNA5 (gfx1250) transformer layer + MoE.
// wave32, WMMA f32<=f16 16x16x32, TDM tensor_load_to_lds for A-tile staging.
// ---------------------------------------------------------------------------

typedef __attribute__((ext_vector_type(16))) _Float16 v16h;
typedef __attribute__((ext_vector_type(8)))  _Float16 v8h;
typedef __attribute__((ext_vector_type(4)))  _Float16 h4v;
typedef __attribute__((ext_vector_type(8)))  float    v8f;
typedef __attribute__((ext_vector_type(4)))  unsigned int u32x4;
typedef __attribute__((ext_vector_type(8)))  int      i32x8;
typedef __attribute__((ext_vector_type(4)))  int      i32x4;

__device__ __forceinline__ v16h cat16(v8h a, v8h b) {
  return __builtin_shufflevector(a, b, 0,1,2,3,4,5,6,7,8,9,10,11,12,13,14,15);
}

__device__ __forceinline__ v8f wmma_f16(v16h a, v16h b, v8f c) {
  // (neg_a, A, neg_b, B, c_mod, C, reuse_a, reuse_b)
  return __builtin_amdgcn_wmma_f32_16x16x32_f16(false, a, false, b, (short)0, c,
                                                false, false);
}

// ---------------------------------------------------------------------------
// f32 -> f16 elementwise convert (vectorized), n % 4 == 0.
// ---------------------------------------------------------------------------
__global__ __launch_bounds__(256)
void cvt_kernel(const float* __restrict__ s, _Float16* __restrict__ d, long n) {
  long i = ((long)blockIdx.x * 256 + threadIdx.x) * 4;
  if (i + 3 < n) {
    float4 v = *(const float4*)(s + i);
    h4v o = { (_Float16)v.x, (_Float16)v.y, (_Float16)v.z, (_Float16)v.w };
    *(h4v*)(d + i) = o;
  }
}

// ---------------------------------------------------------------------------
// LayerNorm (f32 in, f16 out): one 256-thread block per row of D=1024.
// ---------------------------------------------------------------------------
__global__ __launch_bounds__(256)
void ln_kernel(const float* __restrict__ x, const float* __restrict__ g,
               const float* __restrict__ b, _Float16* __restrict__ y, int D) {
  const int row = blockIdx.x;
  const float* xr = x + (size_t)row * D;
  float s = 0.f, ss = 0.f;
  for (int i = threadIdx.x; i < D; i += 256) {
    float v = xr[i];
    s += v; ss += v * v;
  }
  #pragma unroll
  for (int off = 16; off > 0; off >>= 1) {
    s  += __shfl_xor(s,  off, 32);
    ss += __shfl_xor(ss, off, 32);
  }
  __shared__ float r0[8], r1[8];
  const int wave = threadIdx.x >> 5, lane = threadIdx.x & 31;
  if (lane == 0) { r0[wave] = s; r1[wave] = ss; }
  __syncthreads();
  float ts = 0.f, tss = 0.f;
  #pragma unroll
  for (int w = 0; w < 8; ++w) { ts += r0[w]; tss += r1[w]; }
  const float mean = ts / D;
  const float var  = tss / D - mean * mean;
  const float inv  = rsqrtf(var + 1e-5f);
  _Float16* yr = y + (size_t)row * D;
  for (int i = threadIdx.x; i < D; i += 256)
    yr[i] = (_Float16)((xr[i] - mean) * inv * g[i] + b[i]);
}

// ---------------------------------------------------------------------------
// Generic WMMA GEMM:  C[M,N] = op(A[M,K](f16) @ B[K,N](f16) + bias)
//   epilogue: optional residual addsrc (f32), relu, per-row gate, accumulate.
//   output: f16 (Ch != null) or f32 (Cf).
// Block: 256 threads (8 waves), tile 128(M) x 64(N); wave -> 32x32 patch.
// A tile staged by the Tensor Data Mover (tensor_load_to_lds + TENSORcnt);
// B tile staged transposed via b128 loads.
// ---------------------------------------------------------------------------
#define BM 128
#define BN 64
#define BK 32

__global__ __launch_bounds__(256)
void gemm_wmma_kernel(const _Float16* __restrict__ A,
                      const _Float16* __restrict__ B,
                      const float* __restrict__ bias,
                      const float* __restrict__ addsrc,
                      const float* __restrict__ gate, int gateStride,
                      float* __restrict__ Cf, _Float16* __restrict__ Ch,
                      int M, int N, int K, int relu, int accum) {
  __shared__ alignas(16) _Float16 As[BM][BK];   // 8 KB, row-major (TDM dest)
  __shared__ alignas(16) _Float16 Bt[BN][BK];   // 4 KB, B transposed

  const int tid  = threadIdx.x;
  const int wave = tid >> 5;
  const int lane = tid & 31;
  const int wm   = wave >> 1;        // 0..3 -> 32-row band
  const int wn   = wave & 1;         // 0..1 -> 32-col band
  const int hg   = lane >> 4;        // half-group (lanes 16..31)
  const int lr   = lane & 15;

  const int m0 = blockIdx.y * BM;
  const int n0 = blockIdx.x * BN;
  const unsigned int ldsA = (unsigned int)(uintptr_t)(&As[0][0]);

  v8f acc[2][2] = {};

  for (int k0 = 0; k0 < K; k0 += BK) {
    __syncthreads();
#if __has_builtin(__builtin_amdgcn_tensor_load_to_lds)
    if (wave == 0) {
      // ---- TDM: DMA the 128x32 f16 A tile (row stride K) into LDS --------
      const unsigned long long ga =
          (unsigned long long)(uintptr_t)(A + (size_t)m0 * K + k0);
      u32x4 g0 = { 1u,                                  // count=1, user mode
                   ldsA,                                // lds_addr
                   (unsigned int)ga,                    // global_addr[31:0]
                   (unsigned int)((ga >> 32) & 0x1FFFFFFu) | (2u << 30) };
      i32x8 g1 = { (int)(1u << 16),                     // data_size = 2 bytes
                   (int)((K & 0xFFFF) << 16),           // tensor_dim0 lo16
                   (int)(((unsigned)K >> 16) | ((M & 0xFFFF) << 16)),
                   (int)(((unsigned)M >> 16) | (BK << 16)),   // tile_dim0=32
                   BM,                                  // tile_dim1=128
                   K,                                   // dim0_stride lo32
                   0, 0 };
      i32x4 gz = { 0, 0, 0, 0 };
#if defined(__clang_major__) && (__clang_major__ >= 23)
      i32x8 z8 = { 0, 0, 0, 0, 0, 0, 0, 0 };
      __builtin_amdgcn_tensor_load_to_lds(g0, g1, gz, gz, z8, 0);
#else
      __builtin_amdgcn_tensor_load_to_lds(g0, g1, gz, gz, 0);
#endif
      __builtin_amdgcn_s_wait_tensorcnt(0);
    }
#else
    // Fallback: manual b128 staging of A (row-major)
    #pragma unroll
    for (int it = 0; it < 2; ++it) {
      int i = tid + it * 256;                 // 512 x v8h = 4096 halfs
      int r = i >> 2, c = (i & 3) * 8;
      *(v8h*)&As[r][c] = *(const v8h*)&A[(size_t)(m0 + r) * K + k0 + c];
    }
#endif
    // ---- stage B transposed: one b128 load + b16 scatter per thread ------
    {
      const int r = tid >> 3;                 // k within tile (0..31)
      const int c = (tid & 7) * 8;            // n within tile
      v8h bv = *(const v8h*)&B[(size_t)(k0 + r) * N + n0 + c];
      #pragma unroll
      for (int q = 0; q < 8; ++q) Bt[c + q][r] = bv[q];
    }
    if (k0 + BK < K)   // gfx1250 global_prefetch_b8 of next B slice
      __builtin_prefetch(&B[(size_t)(k0 + BK + (tid & 31)) * N + n0], 0, 1);
    __syncthreads();

    // A fragment: lane -> M row, elems 0..7 = K hg*8.., 8..15 = K hg*8+16..
    v16h af[2], bf[2];
    #pragma unroll
    for (int tm = 0; tm < 2; ++tm) {
      const _Float16* p = &As[wm * 32 + tm * 16 + lr][0];
      af[tm] = cat16(*(const v8h*)(p + hg * 8), *(const v8h*)(p + hg * 8 + 16));
    }
    // B fragment: lane -> N col, elems 0..7 = K hg*8.., 8..15 = K hg*8+16..
    #pragma unroll
    for (int tn = 0; tn < 2; ++tn) {
      const _Float16* p = &Bt[wn * 32 + tn * 16 + lr][0];
      bf[tn] = cat16(*(const v8h*)(p + hg * 8), *(const v8h*)(p + hg * 8 + 16));
    }
    #pragma unroll
    for (int tm = 0; tm < 2; ++tm)
      #pragma unroll
      for (int tn = 0; tn < 2; ++tn)
        acc[tm][tn] = wmma_f16(af[tm], bf[tn], acc[tm][tn]);
  }

  // epilogue (C layout: lane -> N col, VGPR j -> M row j + 8*hg)
  #pragma unroll
  for (int tm = 0; tm < 2; ++tm)
    #pragma unroll
    for (int tn = 0; tn < 2; ++tn) {
      const int col = n0 + wn * 32 + tn * 16 + lr;
      const float bval = bias ? bias[col] : 0.f;
      #pragma unroll
      for (int j = 0; j < 8; ++j) {
        const int row = m0 + wm * 32 + tm * 16 + j + 8 * hg;
        float v = acc[tm][tn][j] + bval;
        if (addsrc) v += addsrc[(size_t)row * N + col];
        if (relu)   v = fmaxf(v, 0.f);
        if (gate)   v *= gate[(size_t)row * gateStride];
        const size_t o = (size_t)row * N + col;
        if (Ch)         Ch[o] = (_Float16)v;
        else if (accum) Cf[o] += v;
        else            Cf[o] = v;
      }
    }
}

// ---------------------------------------------------------------------------
// Flash attention: grid (S/128, B*H), block 256.  wave = 16 query rows.
// Q/K/V are f16 [B*S, D] with head offset h*64.  QK^T and P@V via WMMA;
// online softmax with 16-lane shfl butterflies; f16 output.
// ---------------------------------------------------------------------------
__global__ __launch_bounds__(256)
void attn_kernel(const _Float16* __restrict__ Q, const _Float16* __restrict__ K,
                 const _Float16* __restrict__ V, const float* __restrict__ mask,
                 _Float16* __restrict__ O, int S, int D, int H) {
  const int bh = blockIdx.y;
  const int b  = bh / H;
  const int h  = bh % H;
  const int q0 = blockIdx.x * 128;
  const size_t base = (size_t)b * S * D + (size_t)h * 64;

  __shared__ alignas(16) _Float16 Qs[128][64];     // 16 KB
  __shared__ alignas(16) _Float16 Ks[32][64];      //  4 KB
  __shared__ alignas(16) _Float16 Vt[64][32];      //  4 KB (transposed)
  __shared__ alignas(16) _Float16 Pw[8][16][32];   //  8 KB per-wave P bounce
  __shared__ float mk[32];

  const int tid  = threadIdx.x;
  const int wave = tid >> 5;
  const int lane = tid & 31;
  const int hg   = lane >> 4;
  const int lr   = lane & 15;

  #pragma unroll
  for (int it = 0; it < 4; ++it) {                 // 1024 x v8h = 8192 halfs
    int i = tid + it * 256;
    int r = i >> 3, c = (i & 7) * 8;
    *(v8h*)&Qs[r][c] = *(const v8h*)&Q[base + (size_t)(q0 + r) * D + c];
  }

  v8f acc[4] = {};
  float rmax[8], rsum[8];
  #pragma unroll
  for (int j = 0; j < 8; ++j) { rmax[j] = -3.0e38f; rsum[j] = 0.f; }
  const float scale = 0.125f;   // 1/sqrt(64)

  for (int kt = 0; kt < S; kt += 32) {
    __syncthreads();
    {                                              // 256 x v8h = 2048 halfs
      int r = tid >> 3, c = (tid & 7) * 8;
      *(v8h*)&Ks[r][c] = *(const v8h*)&K[base + (size_t)(kt + r) * D + c];
      v8h vv = *(const v8h*)&V[base + (size_t)(kt + r) * D + c];
      #pragma unroll
      for (int q = 0; q < 8; ++q) Vt[c + q][r] = vv[q];
    }
    if (tid < 32) mk[tid] = mask[(size_t)b * S + kt + tid];
    __syncthreads();

    // ---- scores for the two 16-key sub-tiles
    #pragma unroll
    for (int nt = 0; nt < 2; ++nt) {
      v8f sc = {};
      #pragma unroll
      for (int ks = 0; ks < 64; ks += 32) {
        const _Float16* qp = &Qs[wave * 16 + lr][ks];
        v16h aq = cat16(*(const v8h*)(qp + hg * 8), *(const v8h*)(qp + hg * 8 + 16));
        const _Float16* kp = &Ks[nt * 16 + lr][ks];
        v16h bk = cat16(*(const v8h*)(kp + hg * 8), *(const v8h*)(kp + hg * 8 + 16));
        sc = wmma_f16(aq, bk, sc);
      }
      const float addm = (1.0f - mk[nt * 16 + lr]) * -1e9f;
      float pj[8];
      #pragma unroll
      for (int j = 0; j < 8; ++j) {
        float sv = sc[j] * scale + addm;
        float tm = sv;
        #pragma unroll
        for (int off = 1; off < 16; off <<= 1)
          tm = fmaxf(tm, __shfl_xor(tm, off, 32));
        const float nm   = fmaxf(rmax[j], tm);
        const float corr = __expf(rmax[j] - nm);
        float p = __expf(sv - nm);
        float ps = p;
        #pragma unroll
        for (int off = 1; off < 16; off <<= 1)
          ps += __shfl_xor(ps, off, 32);
        rsum[j] = rsum[j] * corr + ps;
        rmax[j] = nm;
        #pragma unroll
        for (int t = 0; t < 4; ++t) acc[t][j] *= corr;
        pj[j] = p;
      }
      // C-layout -> LDS (row m = j + 8*hg, col = key within 32-tile)
      #pragma unroll
      for (int j = 0; j < 8; ++j)
        Pw[wave][j + 8 * hg][nt * 16 + lr] = (_Float16)pj[j];
    }
    // CDNA5 split counters: make the per-wave P stores visible to our loads
    asm volatile("s_wait_dscnt 0x0" ::: "memory");

    // ---- P(16x32) @ V(32x64)
    const _Float16* pp = &Pw[wave][lr][0];
    v16h pf = cat16(*(const v8h*)(pp + hg * 8), *(const v8h*)(pp + hg * 8 + 16));
    #pragma unroll
    for (int t = 0; t < 4; ++t) {
      const _Float16* vp = &Vt[t * 16 + lr][0];
      v16h vf = cat16(*(const v8h*)(vp + hg * 8), *(const v8h*)(vp + hg * 8 + 16));
      acc[t] = wmma_f16(pf, vf, acc[t]);
    }
  }

  #pragma unroll
  for (int t = 0; t < 4; ++t)
    #pragma unroll
    for (int j = 0; j < 8; ++j) {
      const int row = q0 + wave * 16 + j + 8 * hg;
      const int col = t * 16 + lr;
      O[base + (size_t)row * D + col] = (_Float16)(acc[t][j] / rsum[j]);
    }
}

// ---------------------------------------------------------------------------
// Router: softmax over E=8 logits, top-2, renormalize.  One wave per token.
// ---------------------------------------------------------------------------
__global__ __launch_bounds__(256)
void router_kernel(const _Float16* __restrict__ xn, const float* __restrict__ Wr,
                   const float* __restrict__ br, float* __restrict__ gates,
                   int D, int Ntok) {
  const int token = blockIdx.x * 8 + (threadIdx.x >> 5);
  const int lane  = threadIdx.x & 31;
  if (token >= Ntok) return;
  const _Float16* xr = xn + (size_t)token * D;
  float acc[8] = {0.f, 0.f, 0.f, 0.f, 0.f, 0.f, 0.f, 0.f};
  for (int d = lane; d < D; d += 32) {
    const float xv = (float)xr[d];
    const float* wr = Wr + (size_t)d * 8;
    #pragma unroll
    for (int e = 0; e < 8; ++e) acc[e] += xv * wr[e];
  }
  #pragma unroll
  for (int e = 0; e < 8; ++e)
    #pragma unroll
    for (int off = 16; off > 0; off >>= 1)
      acc[e] += __shfl_xor(acc[e], off, 32);
  if (lane == 0) {
    float l[8], mx = -3.0e38f;
    #pragma unroll
    for (int e = 0; e < 8; ++e) { l[e] = acc[e] + br[e]; mx = fmaxf(mx, l[e]); }
    float sum = 0.f;
    #pragma unroll
    for (int e = 0; e < 8; ++e) { l[e] = __expf(l[e] - mx); sum += l[e]; }
    #pragma unroll
    for (int e = 0; e < 8; ++e) l[e] /= sum;
    int i1 = 0;
    #pragma unroll
    for (int e = 1; e < 8; ++e) if (l[e] > l[i1]) i1 = e;
    int i2 = (i1 == 0) ? 1 : 0;
    #pragma unroll
    for (int e = 0; e < 8; ++e) if (e != i1 && l[e] > l[i2]) i2 = e;
    const float t = l[i1] + l[i2];
    float* gp = gates + (size_t)token * 8;
    #pragma unroll
    for (int e = 0; e < 8; ++e) gp[e] = 0.f;
    gp[i1] = l[i1] / t;
    gp[i2] = l[i2] / t;
  }
}

// ---------------------------------------------------------------------------
// Host-side orchestration.
// ---------------------------------------------------------------------------
extern "C" void kernel_launch(void* const* d_in, const int* in_sizes, int n_in,
                              void* d_out, int out_size, void* d_ws, size_t ws_size,
                              hipStream_t stream) {
  (void)in_sizes; (void)n_in; (void)out_size; (void)ws_size;
  const float* x     = (const float*)d_in[0];
  const float* amask = (const float*)d_in[1];
  const float* ln1g  = (const float*)d_in[2];
  const float* ln1b  = (const float*)d_in[3];
  const float* ln2g  = (const float*)d_in[4];
  const float* ln2b  = (const float*)d_in[5];
  const float* Wq = (const float*)d_in[6];  const float* bq = (const float*)d_in[7];
  const float* Wk = (const float*)d_in[8];  const float* bk = (const float*)d_in[9];
  const float* Wv = (const float*)d_in[10]; const float* bv = (const float*)d_in[11];
  const float* Wo = (const float*)d_in[12]; const float* bo = (const float*)d_in[13];
  const float* Wr = (const float*)d_in[14]; const float* br = (const float*)d_in[15];
  const float* W1 = (const float*)d_in[16]; const float* b1 = (const float*)d_in[17];
  const float* W2 = (const float*)d_in[18]; const float* b2 = (const float*)d_in[19];
  float* out = (float*)d_out;

  const int Bb = 4, S = 2048, Dd = 1024, Hh = 16, Ff = 4096, Ee = 8;
  const int NT = Bb * S;                      // 8192 tokens
  typedef _Float16 h16;

  // ---- workspace layout (f32 region first, then f16 region) --------------
  float* wsf   = (float*)d_ws;
  float* xres  = wsf;                         // [NT, D] f32
  float* gates = xres + (size_t)NT * Dd;      // [NT, E] f32
  h16* wsh  = (h16*)(gates + (size_t)NT * Ee);
  h16* xn   = wsh;                            // [NT, D] f16 (ln1, later ln2)
  h16* qb   = xn  + (size_t)NT * Dd;
  h16* kb   = qb  + (size_t)NT * Dd;
  h16* vb   = kb  + (size_t)NT * Dd;
  h16* ab   = vb  + (size_t)NT * Dd;          // attention out, f16
  h16* hbuf = ab  + (size_t)NT * Dd;          // [NT, F] expert hidden, f16
  h16* Wqh  = hbuf + (size_t)NT * Ff;         // f16 weights
  h16* Wkh  = Wqh + (size_t)Dd * Dd;
  h16* Wvh  = Wkh + (size_t)Dd * Dd;
  h16* Woh  = Wvh + (size_t)Dd * Dd;
  h16* W1h  = Woh + (size_t)Dd * Dd;          // [E, D, F]
  h16* W2h  = W1h + (size_t)Ee * Dd * Ff;     // [E, F, D]

  const dim3 blk(256);
  const dim3 gD(Dd / BN, NT / BM);            // N=1024 outputs
  const dim3 gF(Ff / BN, NT / BM);            // N=4096 outputs

  // ---- convert weights to f16 --------------------------------------------
  const long nDD = (long)Dd * Dd, nEDF = (long)Ee * Dd * Ff;
  cvt_kernel<<<(unsigned)(nDD / 1024),  blk, 0, stream>>>(Wq, Wqh, nDD);
  cvt_kernel<<<(unsigned)(nDD / 1024),  blk, 0, stream>>>(Wk, Wkh, nDD);
  cvt_kernel<<<(unsigned)(nDD / 1024),  blk, 0, stream>>>(Wv, Wvh, nDD);
  cvt_kernel<<<(unsigned)(nDD / 1024),  blk, 0, stream>>>(Wo, Woh, nDD);
  cvt_kernel<<<(unsigned)(nEDF / 1024), blk, 0, stream>>>(W1, W1h, nEDF);
  cvt_kernel<<<(unsigned)(nEDF / 1024), blk, 0, stream>>>(W2, W2h, nEDF);

  // ---- LN1 + QKV projections ---------------------------------------------
  ln_kernel<<<NT, blk, 0, stream>>>(x, ln1g, ln1b, xn, Dd);
  gemm_wmma_kernel<<<gD, blk, 0, stream>>>(xn, Wqh, bq, nullptr, nullptr, 0,
                                           nullptr, qb, NT, Dd, Dd, 0, 0);
  gemm_wmma_kernel<<<gD, blk, 0, stream>>>(xn, Wkh, bk, nullptr, nullptr, 0,
                                           nullptr, kb, NT, Dd, Dd, 0, 0);
  gemm_wmma_kernel<<<gD, blk, 0, stream>>>(xn, Wvh, bv, nullptr, nullptr, 0,
                                           nullptr, vb, NT, Dd, Dd, 0, 0);
  // ---- flash attention ----------------------------------------------------
  attn_kernel<<<dim3(S / 128, Bb * Hh), blk, 0, stream>>>(qb, kb, vb, amask,
                                                          ab, S, Dd, Hh);
  // ---- output projection + residual (xres = x + attn @ Wo + bo) ----------
  gemm_wmma_kernel<<<gD, blk, 0, stream>>>(ab, Woh, bo, x, nullptr, 0,
                                           xres, nullptr, NT, Dd, Dd, 0, 0);
  // ---- LN2 + router -------------------------------------------------------
  ln_kernel<<<NT, blk, 0, stream>>>(xres, ln2g, ln2b, xn, Dd);
  router_kernel<<<NT / 8, blk, 0, stream>>>(xn, Wr, br, gates, Dd, NT);
  // out starts as residual stream; experts accumulate into it
  (void)hipMemcpyAsync(out, xres, (size_t)NT * Dd * sizeof(float),
                       hipMemcpyDeviceToDevice, stream);
  for (int e = 0; e < Ee; ++e) {
    const h16*   W1e = W1h + (size_t)e * Dd * Ff;
    const float* b1e = b1  + (size_t)e * Ff;
    const h16*   W2e = W2h + (size_t)e * Ff * Dd;
    const float* b2e = b2  + (size_t)e * Dd;
    // h = relu(xn @ W1[e] + b1[e])  -> f16
    gemm_wmma_kernel<<<gF, blk, 0, stream>>>(xn, W1e, b1e, nullptr, nullptr, 0,
                                             nullptr, hbuf, NT, Ff, Dd, 1, 0);
    // out += gate[:,e] * (h @ W2[e] + b2[e])
    gemm_wmma_kernel<<<gD, blk, 0, stream>>>(hbuf, W2e, b2e, nullptr,
                                             gates + e, Ee,
                                             out, nullptr, NT, Dd, Ff, 0, 1);
  }
}
